// MambaCell_65704409694785
// MI455X (gfx1250) — matrix operand order repaired and checked by reading, not compiled, not run on hardware
//
#include <hip/hip_runtime.h>
#include <hip/hip_bf16.h>
#include <math.h>

// Problem constants (match reference)
constexpr int Bb = 4;
constexpr int L  = 1024;
constexpr int DM = 1024;   // D_MODEL
constexpr int DI = 2048;   // D_INNER
constexpr int DS = 16;     // D_STATE
constexpr int DC = 4;      // D_CONV
constexpr int DR = 64;     // DT_RANK
constexpr int T  = Bb * L; // 4096 token rows
constexpr int XD = DR + 2 * DS; // 96

typedef __attribute__((ext_vector_type(16))) _Float16 v16h;
typedef __attribute__((ext_vector_type(8)))  float    v8f;

union Frag16 { v16h v; uint4 u[2]; };

// ---------------------------------------------------------------------------
// LayerNorm over DM, writes f16 activations for WMMA + fp32 residual copy.
// ---------------------------------------------------------------------------
__global__ __launch_bounds__(256)
void ln_kernel(const float* __restrict__ x, const float* __restrict__ w,
               const float* __restrict__ b, _Float16* __restrict__ h,
               float* __restrict__ resid)
{
    __shared__ float s1[256], s2[256];
    const int row = blockIdx.x;
    const int tid = threadIdx.x;
    const float* xr = x + (size_t)row * DM;
    float s = 0.f, ss = 0.f;
    for (int i = tid; i < DM; i += 256) { float v = xr[i]; s += v; ss += v * v; }
    s1[tid] = s; s2[tid] = ss;
    __syncthreads();
    for (int st = 128; st > 0; st >>= 1) {
        if (tid < st) { s1[tid] += s1[tid + st]; s2[tid] += s2[tid + st]; }
        __syncthreads();
    }
    const float mean = s1[0] * (1.f / DM);
    const float var  = s2[0] * (1.f / DM) - mean * mean;
    const float rstd = rsqrtf(var + 1e-5f);
    for (int i = tid; i < DM; i += 256) {
        float v = xr[i];
        resid[(size_t)row * DM + i] = v;
        h[(size_t)row * DM + i] = (_Float16)((v - mean) * rstd * w[i] + b[i]);
    }
}

// ---------------------------------------------------------------------------
// fp32 -> f16 cast (weights)
// ---------------------------------------------------------------------------
__global__ __launch_bounds__(256)
void cast_f32_f16_kernel(const float* __restrict__ src, _Float16* __restrict__ dst, int n)
{
    int i = blockIdx.x * 256 + threadIdx.x;
    if (i < n) dst[i] = (_Float16)src[i];
}

// Strided cast of dt-rank columns out of x_dbl [T x 96] -> [T x 64] f16
__global__ __launch_bounds__(256)
void cast_dt_kernel(const float* __restrict__ xdbl, _Float16* __restrict__ dtp)
{
    int i = blockIdx.x * 256 + threadIdx.x; // T*DR
    int t = i >> 6, r = i & 63;
    dtp[i] = (_Float16)xdbl[(size_t)t * XD + r];
}

// ---------------------------------------------------------------------------
// WMMA GEMM:  C[M x N] (f32) = A[M x K] (f16, row-major) * W[N x K]^T (f16)
// EPI==1: C = softplus(C + bias[n])
// Per wave: 64(M) x 32(N) output -> 4x2 tiles of v_wmma_f32_16x16x32_f16
//           (8 WMMA per k-step for 12 b128 fragment loads: ~11 flops/byte).
// Block: 8 waves arranged 2(M) x 4(N) -> 128 x 128 block tile.
// ---------------------------------------------------------------------------
template <int EPI>
__global__ __launch_bounds__(256)
void wmma_gemm_kernel(const _Float16* __restrict__ A,
                      const _Float16* __restrict__ W,
                      float* __restrict__ C,
                      int M, int N, int K,
                      const float* __restrict__ bias)
{
    const int lane  = threadIdx.x & 31;
    const int wave  = threadIdx.x >> 5;
    const int mwave = wave & 1;          // 2 wave rows in M
    const int nwave = wave >> 1;         // 4 wave cols in N
    const int m0 = blockIdx.x * 128 + mwave * 64;
    const int n0 = blockIdx.y * 128 + nwave * 32;
    if (n0 >= N) return;                 // wave-uniform skip (EXEC stays all-1)

    const int mrow  = lane & 15;
    const int hi    = lane >> 4;         // 0 or 1
    const int abase = hi * 8;            // A-frag: K offsets {abase..+7, abase+16..+23}
    const int bbase = hi * 8;            // B-frag: K offsets {bbase..+7, bbase+16..+23}
    const int ncol  = lane & 15;

    v8f acc[4][2];
#pragma unroll
    for (int i = 0; i < 4; ++i)
#pragma unroll
        for (int j = 0; j < 2; ++j)
            acc[i][j] = (v8f){};

    // Per-lane clamped B rows (keeps loads in-bounds when N not multiple of 32)
    int nrow0 = n0 + ncol;      if (nrow0 > N - 1) nrow0 = N - 1;
    int nrow1 = n0 + 16 + ncol; if (nrow1 > N - 1) nrow1 = N - 1;

    const _Float16* ap[4];
    ap[0] = A + (size_t)(m0 + mrow) * K + abase;
    ap[1] = ap[0] + (size_t)16 * K;
    ap[2] = ap[0] + (size_t)32 * K;
    ap[3] = ap[0] + (size_t)48 * K;
    const _Float16* b0p = W + (size_t)nrow0 * K + bbase;
    const _Float16* b1p = W + (size_t)nrow1 * K + bbase;

    for (int k0 = 0; k0 < K; k0 += 32) {
        Frag16 af[4], b0, b1;
#pragma unroll
        for (int i = 0; i < 4; ++i) {
            af[i].u[0] = *(const uint4*)(ap[i] + k0);
            af[i].u[1] = *(const uint4*)(ap[i] + k0 + 16);
        }
        b0.u[0] = *(const uint4*)(b0p + k0);
        b0.u[1] = *(const uint4*)(b0p + k0 + 16);
        b1.u[0] = *(const uint4*)(b1p + k0);
        b1.u[1] = *(const uint4*)(b1p + k0 + 16);

#pragma unroll
        for (int i = 0; i < 4; ++i) {
            acc[i][0] = __builtin_amdgcn_wmma_f32_16x16x32_f16(
                false, af[i].v, false, b0.v, (short)0, acc[i][0], false, false);
            acc[i][1] = __builtin_amdgcn_wmma_f32_16x16x32_f16(
                false, af[i].v, false, b1.v, (short)0, acc[i][1], false, false);
        }
    }

    // C/D layout: lanes 0-15 -> N=lane, M = vgpr r; lanes 16-31 -> M = r + 8
#pragma unroll
    for (int i = 0; i < 4; ++i) {
        const int mb = m0 + i * 16 + hi * 8;
#pragma unroll
        for (int j = 0; j < 2; ++j) {
            const int col = n0 + j * 16 + ncol;
            if (col < N) {
                float bv = (EPI == 1) ? bias[col] : 0.f;
#pragma unroll
                for (int r = 0; r < 8; ++r) {
                    float v = acc[i][j][r];
                    if (EPI == 1) {
                        v += bv;
                        v = (v > 20.f) ? v : log1pf(__expf(v));
                    }
                    C[(size_t)(mb + r) * N + col] = v;
                }
            }
        }
    }
}

// ---------------------------------------------------------------------------
// Depthwise causal conv (width 4) + SiLU on x half of xz; writes f32 + f16.
// ---------------------------------------------------------------------------
__global__ __launch_bounds__(256)
void conv_silu_kernel(const float* __restrict__ xz, const float* __restrict__ cw,
                      const float* __restrict__ cb, float* __restrict__ xs_f,
                      _Float16* __restrict__ xs_h)
{
    size_t i = (size_t)blockIdx.x * 256 + threadIdx.x; // over T*DI
    const int d = (int)(i & (DI - 1));
    const int g = (int)(i >> 11);       // token row (b*L + l)
    const int l = g & (L - 1);
    float acc = cb[d];
#pragma unroll
    for (int k = 0; k < DC; ++k) {
        int ls = l + k - (DC - 1);
        if (ls >= 0)
            acc += xz[(size_t)(g + k - (DC - 1)) * (2 * DI) + d] * cw[d * DC + k];
    }
    float sv = acc / (1.f + __expf(-acc));   // silu
    xs_f[i] = sv;
    xs_h[i] = (_Float16)sv;
}

// ---------------------------------------------------------------------------
// Selective scan: one lane per (b, d); 16-element state in registers.
// Fuses + x*D_skip and * silu(z); emits f16 for the output GEMM.
// ---------------------------------------------------------------------------
__global__ __launch_bounds__(256)
void scan_kernel(const float* __restrict__ dt, const float* __restrict__ xs,
                 const float* __restrict__ xdbl, const float* __restrict__ xz,
                 const float* __restrict__ A_log, const float* __restrict__ D_skip,
                 _Float16* __restrict__ y_h)
{
    const int idx = blockIdx.x * 256 + threadIdx.x;   // 0 .. Bb*DI-1
    const int b = idx >> 11;          // DI == 2048
    const int d = idx & (DI - 1);

    float a[DS], h[DS];
#pragma unroll
    for (int n = 0; n < DS; ++n) {
        a[n] = -__expf(A_log[(size_t)d * DS + n]);
        h[n] = 0.f;
    }
    const float dsk = D_skip[d];

    for (int l = 0; l < L; ++l) {
        const size_t g = (size_t)b * L + l;
        const float dtv = dt[g * DI + d];
        const float xv  = xs[g * DI + d];
        const float zv  = xz[g * (2 * DI) + DI + d];
        const float* row = xdbl + g * XD;
        const float dx = dtv * xv;
        float y = 0.f;
#pragma unroll
        for (int n = 0; n < DS; ++n) {
            const float Bn = row[DR + n];
            const float Cn = row[DR + DS + n];
            h[n] = __expf(dtv * a[n]) * h[n] + dx * Bn;
            y += h[n] * Cn;
        }
        y += xv * dsk;
        y *= zv / (1.f + __expf(-zv));    // * silu(z)
        y_h[g * DI + d] = (_Float16)y;
    }
}

// ---------------------------------------------------------------------------
extern "C" void kernel_launch(void* const* d_in, const int* in_sizes, int n_in,
                              void* d_out, int out_size, void* d_ws, size_t ws_size,
                              hipStream_t stream)
{
    const float* x      = (const float*)d_in[0];
    const float* ln_w   = (const float*)d_in[1];
    const float* ln_b   = (const float*)d_in[2];
    const float* W_in   = (const float*)d_in[3];
    const float* conv_w = (const float*)d_in[4];
    const float* conv_b = (const float*)d_in[5];
    const float* W_x    = (const float*)d_in[6];
    const float* W_dt   = (const float*)d_in[7];
    const float* b_dt   = (const float*)d_in[8];
    const float* A_log  = (const float*)d_in[9];
    const float* D_skip = (const float*)d_in[10];
    const float* W_out  = (const float*)d_in[11];

    float* out      = (float*)d_out;                    // [T, DM]
    float* residual = out + (size_t)T * DM;             // [T, DM]

    // Workspace carve-out
    char* ws = (char*)d_ws;
    size_t off = 0;
    auto carve = [&](size_t bytes) -> char* {
        char* p = ws + off;
        off += (bytes + 255) & ~(size_t)255;
        return p;
    };
    _Float16* h_h    = (_Float16*)carve((size_t)T * DM * 2);        // LN output f16
    _Float16* Win_h  = (_Float16*)carve((size_t)2 * DI * DM * 2);
    _Float16* Wx_h   = (_Float16*)carve((size_t)XD * DI * 2);
    _Float16* Wdt_h  = (_Float16*)carve((size_t)DI * DR * 2);
    _Float16* Wout_h = (_Float16*)carve((size_t)DM * DI * 2);
    float*    xz     = (float*)carve((size_t)T * 2 * DI * 4);       // [T, 4096]
    float*    xs_f   = (float*)carve((size_t)T * DI * 4);           // silu(conv(x)) f32
    _Float16* xs_h   = (_Float16*)carve((size_t)T * DI * 2);        // f16 (later reused for y)
    float*    xdbl   = (float*)carve((size_t)T * XD * 4);           // [T, 96]
    _Float16* dtp_h  = (_Float16*)carve((size_t)T * DR * 2);
    float*    dt_f   = (float*)carve((size_t)T * DI * 4);           // softplus dt
    _Float16* y_h    = xs_h; // xs_h no longer needed after GEMM2 -> reuse for y

    const dim3 blk(256);

    // 1) LayerNorm (+ residual copy)
    ln_kernel<<<T, blk, 0, stream>>>(x, ln_w, ln_b, h_h, residual);

    // 2) Weight casts to f16
    cast_f32_f16_kernel<<<(2 * DI * DM + 255) / 256, blk, 0, stream>>>(W_in, Win_h, 2 * DI * DM);
    cast_f32_f16_kernel<<<(XD * DI + 255) / 256, blk, 0, stream>>>(W_x, Wx_h, XD * DI);
    cast_f32_f16_kernel<<<(DI * DR + 255) / 256, blk, 0, stream>>>(W_dt, Wdt_h, DI * DR);
    cast_f32_f16_kernel<<<(DM * DI + 255) / 256, blk, 0, stream>>>(W_out, Wout_h, DM * DI);

    // 3) xz = h @ W_in^T   [4096 x 4096 x 1024]
    wmma_gemm_kernel<0><<<dim3(T / 128, (2 * DI) / 128), blk, 0, stream>>>(
        h_h, Win_h, xz, T, 2 * DI, DM, nullptr);

    // 4) x = silu(causal depthwise conv(x))
    conv_silu_kernel<<<(T * DI) / 256, blk, 0, stream>>>(xz, conv_w, conv_b, xs_f, xs_h);

    // 5) x_dbl = x @ W_x^T   [4096 x 96 x 2048]
    wmma_gemm_kernel<0><<<dim3(T / 128, (XD + 127) / 128), blk, 0, stream>>>(
        xs_h, Wx_h, xdbl, T, XD, DI, nullptr);

    // 6) dt = softplus(dt_rank @ W_dt^T + b_dt)   [4096 x 2048 x 64]
    cast_dt_kernel<<<(T * DR) / 256, blk, 0, stream>>>(xdbl, dtp_h);
    wmma_gemm_kernel<1><<<dim3(T / 128, DI / 128), blk, 0, stream>>>(
        dtp_h, Wdt_h, dt_f, T, DI, DR, b_dt);

    // 7) selective scan + D-skip + silu(z) gate -> y (f16)
    scan_kernel<<<(Bb * DI) / 256, blk, 0, stream>>>(
        dt_f, xs_f, xdbl, xz, A_log, D_skip, y_h);

    // 8) out = y @ W_out^T   [4096 x 1024 x 2048]
    wmma_gemm_kernel<0><<<dim3(T / 128, DM / 128), blk, 0, stream>>>(
        y_h, Wout_h, out, T, DM, DI, nullptr);
}
